// EGNNetwork_20298015441436
// MI455X (gfx1250) — matrix operand order repaired and checked by reading, not compiled, not run on hardware
//
#include <hip/hip_runtime.h>
#include <cstdint>
#include <cstddef>

typedef __attribute__((ext_vector_type(16))) _Float16 v16h;
typedef __attribute__((ext_vector_type(8)))  float    v8f;

union Half16 { v16h v; uint4 q[2]; };
union HalfPack4 { _Float16 h[4]; uint2 u; };

__device__ __forceinline__ float fast_rcp(float x) { return __builtin_amdgcn_rcpf(x); }

__device__ __forceinline__ float silu_f(float x) {
  // x * sigmoid(x) with fast (TRANS-pipe) reciprocal: v_exp_f32 + v_rcp_f32
  return x * fast_rcp(1.0f + __expf(-x));
}

__device__ __forceinline__ v8f wmma_f16(v16h a, v16h b, v8f c) {
  // D = A(16x32 f16) * B(32x16 f16) + C(16x16 f32)
  return __builtin_amdgcn_wmma_f32_16x16x32_f16(false, a, false, b, (short)0, c,
                                                false, false);
}

// ---------------- elementwise utility kernels ----------------
__global__ void k_zero_f32(float* __restrict__ p, int n) {
  int i = blockIdx.x * blockDim.x + threadIdx.x;
  int st = gridDim.x * blockDim.x;
  for (; i < n; i += st) p[i] = 0.0f;
}

__global__ void k_f32_to_f16(const float* __restrict__ in, _Float16* __restrict__ out, int n) {
  int i = blockIdx.x * blockDim.x + threadIdx.x;
  int st = gridDim.x * blockDim.x;
  for (; i < n; i += st) out[i] = (_Float16)in[i];
}

__global__ void k_add_f32(const float* __restrict__ a, const float* __restrict__ b,
                          float* __restrict__ o, int n) {
  int i = blockIdx.x * blockDim.x + threadIdx.x;
  int st = gridDim.x * blockDim.x;
  for (; i < n; i += st) o[i] = a[i] + b[i];
}

// W: [K][128] row-major f32  ->  WT: [128][KP] f16 (zero padded K..KP)
__global__ void k_transpose_w(const float* __restrict__ W, _Float16* __restrict__ WT,
                              int K, int KP) {
  int i = blockIdx.x * blockDim.x + threadIdx.x;
  int total = 128 * KP;
  int st = gridDim.x * blockDim.x;
  for (; i < total; i += st) {
    int n = i / KP;
    int k = i - n * KP;
    float v = (k < K) ? W[(size_t)k * 128 + n] : 0.0f;
    WT[(size_t)n * KP + k] = (_Float16)v;
  }
}

// ---------------- fused edge kernel ----------------
// One wave processes a tile of 16 edges:
//   f[16,KP] = [h16[src] | h16[dst] | radial | 0-pad]
//   m1 = silu(f @ We1 + be1); m = silu(m1 @ We2 + be2)  -> atomic scatter to h_nb[dst]
//   q  = silu(m @ Wc1 + bc1); c = q @ Wc2               -> atomic scatter c*x_diff to x_nb[dst]
template <int IN, int KP>
__global__ __launch_bounds__(128) void k_egnn_edge(
    const _Float16* __restrict__ h16, const float* __restrict__ x,
    const int* __restrict__ src, const int* __restrict__ dst, int E,
    const _Float16* __restrict__ We1T, const float* __restrict__ be1,
    const _Float16* __restrict__ We2T, const float* __restrict__ be2,
    const _Float16* __restrict__ Wc1T, const float* __restrict__ bc1,
    const float* __restrict__ Wc2,
    float* __restrict__ h_nb, float* __restrict__ x_nb) {
  constexpr int WV = 4;  // waves per block
  __shared__ __align__(16) _Float16 sT0[WV][16 * KP];
  __shared__ __align__(16) _Float16 sT1[WV][16 * 128];
  __shared__ float sXD[WV][16][3];
  __shared__ float sRAD[WV][16];
  __shared__ int sS[WV][16];
  __shared__ int sD[WV][16];

  const int lane = threadIdx.x & 31;
  const int wid = threadIdx.x >> 5;
  const int gw = blockIdx.x * WV + wid;
  const int nw = gridDim.x * WV;
  const int tiles = (E + 15) >> 4;

  _Float16* T0 = sT0[wid];
  _Float16* T1 = sT1[wid];

  const int colN = lane & 15;   // C/D: column; A: row M
  const int hi = lane >> 4;     // half-wave select
  const int selA = hi * 8;      // A frag half-offset (ISA 16-bit A layout)
  const int selB = hi * 8;      // B frag half-offset (same as A)
  const int myM = hi * 8;       // C/D row base

  for (int tile = gw; tile < tiles; tile += nw) {
    const int ebase = tile << 4;
    // phase 1: per-edge geometry (lanes 0..15)
    if (lane < 16) {
      int e = ebase + lane;
      int valid = e < E;
      int ec = valid ? e : 0;
      int s = src[ec], d = dst[ec];
      float dx = x[s * 3 + 0] - x[d * 3 + 0];
      float dy = x[s * 3 + 1] - x[d * 3 + 1];
      float dz = x[s * 3 + 2] - x[d * 3 + 2];
      float r2 = dx * dx + dy * dy + dz * dz;
      float inv = fast_rcp(sqrtf(r2) + 1e-30f);
      sXD[wid][lane][0] = dx * inv;
      sXD[wid][lane][1] = dy * inv;
      sXD[wid][lane][2] = dz * inv;
      sRAD[wid][lane] = r2;
      sS[wid][lane] = s;
      sD[wid][lane] = valid ? d : -1;
    }
    __builtin_amdgcn_wave_barrier();

    // phase 2: build f tile [16][KP] (f16) in LDS
    for (int e = 0; e < 16; ++e) {
      int s = sS[wid][e];
      int d = sD[wid][e];
      float rad = sRAD[wid][e];
      const _Float16* hs = h16 + (size_t)s * IN;
      const _Float16* hd = h16 + (size_t)(d < 0 ? 0 : d) * IN;
      bool valid = d >= 0;
      if constexpr ((IN % 8) == 0) {
        // vector path: 8 halves per chunk; 2*IN/8 chunks cover [h_src|h_dst]
        constexpr int C = IN / 8;
        const uint4* hs4 = (const uint4*)hs;
        const uint4* hd4 = (const uint4*)hd;
        uint4* t4 = (uint4*)(T0 + e * KP);
        for (int c = lane; c < 2 * C; c += 32) {
          uint4 v = make_uint4(0u, 0u, 0u, 0u);
          if (valid) v = (c < C) ? hs4[c] : hd4[c - C];
          t4[c] = v;
        }
        for (int k = 2 * IN + lane; k < KP; k += 32) {
          _Float16 v = (_Float16)0.0f;
          if (valid && k == 2 * IN) v = (_Float16)rad;
          T0[e * KP + k] = v;
        }
      } else {
        for (int k = lane; k < KP; k += 32) {  // KP % 32 == 0: uniform trip count
          _Float16 v = (_Float16)0.0f;
          if (valid) {
            if (k < IN) v = hs[k];
            else if (k < 2 * IN) v = hd[k - IN];
            else if (k == 2 * IN) v = (_Float16)rad;
          }
          T0[e * KP + k] = v;
        }
      }
    }
    __builtin_amdgcn_wave_barrier();

    int dstM[8];
#pragma unroll
    for (int r = 0; r < 8; ++r) dstM[r] = sD[wid][myM + r];

    // ---- GEMM1: m1 = silu(f @ We1 + be1) -> T1 (f16)
    for (int nb = 0; nb < 8; ++nb) {
      int N = nb * 16 + colN;
      float bv = be1[N];
      v8f acc = {bv, bv, bv, bv, bv, bv, bv, bv};
#pragma unroll
      for (int ks = 0; ks < KP / 32; ++ks) {
        Half16 A, B;
        const uint4* ap = (const uint4*)(T0 + colN * KP + ks * 32 + selA);
        A.q[0] = ap[0];
        A.q[1] = ap[2];  // +16 halves
        const uint4* bp = (const uint4*)(We1T + (size_t)N * KP + ks * 32 + selB);
        B.q[0] = bp[0];
        B.q[1] = bp[2];
        acc = wmma_f16(A.v, B.v, acc);
      }
#pragma unroll
      for (int r = 0; r < 8; ++r) T1[(myM + r) * 128 + N] = (_Float16)silu_f(acc[r]);
    }
    __builtin_amdgcn_wave_barrier();

    // ---- GEMM2: m = silu(m1 @ We2 + be2); scatter to h_nb; keep f16 m in T0
    for (int nb = 0; nb < 8; ++nb) {
      int N = nb * 16 + colN;
      float bv = be2[N];
      v8f acc = {bv, bv, bv, bv, bv, bv, bv, bv};
#pragma unroll
      for (int ks = 0; ks < 4; ++ks) {
        Half16 A, B;
        const uint4* ap = (const uint4*)(T1 + colN * 128 + ks * 32 + selA);
        A.q[0] = ap[0];
        A.q[1] = ap[2];
        const uint4* bp = (const uint4*)(We2T + (size_t)N * 128 + ks * 32 + selB);
        B.q[0] = bp[0];
        B.q[1] = bp[2];
        acc = wmma_f16(A.v, B.v, acc);
      }
#pragma unroll
      for (int r = 0; r < 8; ++r) {
        float mv = silu_f(acc[r]);
        int d = dstM[r];
        if (d >= 0) atomicAdd(&h_nb[(size_t)d * 128 + N], mv);
        T0[(myM + r) * 128 + N] = (_Float16)mv;
      }
    }
    __builtin_amdgcn_wave_barrier();

    // ---- GEMM3: q = silu(m @ Wc1 + bc1); c = q @ Wc2 (per-lane dot + shfl reduce)
    float cp[8];
#pragma unroll
    for (int r = 0; r < 8; ++r) cp[r] = 0.0f;

    for (int nb = 0; nb < 8; ++nb) {
      int N = nb * 16 + colN;
      float bv = bc1[N];
      float w2 = Wc2[N];
      v8f acc = {bv, bv, bv, bv, bv, bv, bv, bv};
#pragma unroll
      for (int ks = 0; ks < 4; ++ks) {
        Half16 A, B;
        const uint4* ap = (const uint4*)(T0 + colN * 128 + ks * 32 + selA);
        A.q[0] = ap[0];
        A.q[1] = ap[2];
        const uint4* bp = (const uint4*)(Wc1T + (size_t)N * 128 + ks * 32 + selB);
        B.q[0] = bp[0];
        B.q[1] = bp[2];
        acc = wmma_f16(A.v, B.v, acc);
      }
#pragma unroll
      for (int r = 0; r < 8; ++r) cp[r] += silu_f(acc[r]) * w2;
    }
    // reduce over the 16 column-lanes of each half-wave
#pragma unroll
    for (int m = 1; m <= 8; m <<= 1) {
#pragma unroll
      for (int r = 0; r < 8; ++r) cp[r] += __shfl_xor(cp[r], m, 32);
    }
    // scatter msg_x = c * x_diff (one lane per edge)
#pragma unroll
    for (int r = 0; r < 8; ++r) {
      if (colN == r) {
        int e = myM + r;
        int d = dstM[r];
        if (d >= 0) {
          float c = cp[r];
          atomicAdd(&x_nb[(size_t)d * 3 + 0], c * sXD[wid][e][0]);
          atomicAdd(&x_nb[(size_t)d * 3 + 1], c * sXD[wid][e][1]);
          atomicAdd(&x_nb[(size_t)d * 3 + 2], c * sXD[wid][e][2]);
        }
      }
    }
  }
}

// ---------------- fused node kernel ----------------
// h_out = silu([h | h_nb] @ Wn1 + bn1) @ Wn2 + bn2   (16-node tiles, in-place safe)
template <int IN, int KP>
__global__ __launch_bounds__(128) void k_egnn_node(
    const float* __restrict__ h_in, const float* __restrict__ h_nb,
    const _Float16* __restrict__ Wn1T, const float* __restrict__ bn1,
    const _Float16* __restrict__ Wn2T, const float* __restrict__ bn2,
    float* __restrict__ h_out, int Nn) {
  constexpr int WV = 4;
  __shared__ __align__(16) _Float16 sT0[WV][16 * KP];
  __shared__ __align__(16) _Float16 sT1[WV][16 * 128];

  const int lane = threadIdx.x & 31;
  const int wid = threadIdx.x >> 5;
  const int gw = blockIdx.x * WV + wid;
  const int nw = gridDim.x * WV;
  const int tiles = (Nn + 15) >> 4;

  _Float16* T0 = sT0[wid];
  _Float16* T1 = sT1[wid];

  const int colN = lane & 15;
  const int hi = lane >> 4;
  const int selA = hi * 8;
  const int selB = hi * 8;
  const int myM = hi * 8;

  for (int tile = gw; tile < tiles; tile += nw) {
    const int nbase = tile << 4;
    // build [h | h_neigh] tile (f16)
    for (int e = 0; e < 16; ++e) {
      int node = nbase + e;
      bool valid = node < Nn;
      int nc = valid ? node : 0;
      const float* hr = h_in + (size_t)nc * IN;
      const float* gr = h_nb + (size_t)nc * 128;
      if constexpr ((IN % 4) == 0) {
        // vector path: float4 loads, packed 8B LDS stores
        constexpr int C1 = IN / 4;
        constexpr int CT = C1 + 32;  // + 128/4 chunks of h_neigh
        const float4* hr4 = (const float4*)hr;
        const float4* gr4 = (const float4*)gr;
        uint2* t2 = (uint2*)(T0 + e * KP);
        for (int c = lane; c < CT; c += 32) {
          float4 v = make_float4(0.f, 0.f, 0.f, 0.f);
          if (valid) v = (c < C1) ? hr4[c] : gr4[c - C1];
          HalfPack4 pk;
          pk.h[0] = (_Float16)v.x;
          pk.h[1] = (_Float16)v.y;
          pk.h[2] = (_Float16)v.z;
          pk.h[3] = (_Float16)v.w;
          t2[c] = pk.u;
        }
        for (int k = IN + 128 + lane; k < KP; k += 32) T0[e * KP + k] = (_Float16)0.0f;
      } else {
        for (int k = lane; k < KP; k += 32) {
          float v = 0.0f;
          if (valid) {
            if (k < IN) v = hr[k];
            else if (k < IN + 128) v = gr[k - IN];
          }
          T0[e * KP + k] = (_Float16)v;
        }
      }
    }
    __builtin_amdgcn_wave_barrier();

    // GEMM1 -> T1
    for (int nb = 0; nb < 8; ++nb) {
      int N = nb * 16 + colN;
      float bv = bn1[N];
      v8f acc = {bv, bv, bv, bv, bv, bv, bv, bv};
#pragma unroll
      for (int ks = 0; ks < KP / 32; ++ks) {
        Half16 A, B;
        const uint4* ap = (const uint4*)(T0 + colN * KP + ks * 32 + selA);
        A.q[0] = ap[0];
        A.q[1] = ap[2];
        const uint4* bp = (const uint4*)(Wn1T + (size_t)N * KP + ks * 32 + selB);
        B.q[0] = bp[0];
        B.q[1] = bp[2];
        acc = wmma_f16(A.v, B.v, acc);
      }
#pragma unroll
      for (int r = 0; r < 8; ++r) T1[(myM + r) * 128 + N] = (_Float16)silu_f(acc[r]);
    }
    __builtin_amdgcn_wave_barrier();

    // GEMM2 -> h_out (f32)
    for (int nb = 0; nb < 8; ++nb) {
      int N = nb * 16 + colN;
      float bv = bn2[N];
      v8f acc = {bv, bv, bv, bv, bv, bv, bv, bv};
#pragma unroll
      for (int ks = 0; ks < 4; ++ks) {
        Half16 A, B;
        const uint4* ap = (const uint4*)(T1 + colN * 128 + ks * 32 + selA);
        A.q[0] = ap[0];
        A.q[1] = ap[2];
        const uint4* bp = (const uint4*)(Wn2T + (size_t)N * 128 + ks * 32 + selB);
        B.q[0] = bp[0];
        B.q[1] = bp[2];
        acc = wmma_f16(A.v, B.v, acc);
      }
#pragma unroll
      for (int r = 0; r < 8; ++r) {
        int node = nbase + myM + r;
        if (node < Nn) h_out[(size_t)node * 128 + N] = acc[r];
      }
    }
  }
}

// ---------------- final mean over nodes ----------------
__global__ void k_mean(const float* __restrict__ h, const float* __restrict__ x,
                       float* __restrict__ out, int Nn) {
  __shared__ float red[256];
  int f = blockIdx.x;  // 0..130
  float s = 0.0f;
  if (f < 128) {
    for (int i = threadIdx.x; i < Nn; i += 256) s += h[(size_t)i * 128 + f];
  } else {
    int d = f - 128;
    for (int i = threadIdx.x; i < Nn; i += 256) s += x[(size_t)i * 3 + d];
  }
  red[threadIdx.x] = s;
  __syncthreads();
  for (int off = 128; off > 0; off >>= 1) {
    if ((int)threadIdx.x < off) red[threadIdx.x] += red[threadIdx.x + off];
    __syncthreads();
  }
  if (threadIdx.x == 0) out[f] = red[0] / (float)Nn;
}

extern "C" void kernel_launch(void* const* d_in, const int* in_sizes, int n_in,
                              void* d_out, int out_size, void* d_ws, size_t ws_size,
                              hipStream_t stream) {
  (void)n_in;
  (void)ws_size;
  const float* h0 = (const float*)d_in[0];
  const float* x0 = (const float*)d_in[1];
  const int* src = (const int*)d_in[2];
  const int* dst = (const int*)d_in[3];
  const int E = in_sizes[2];
  const int Nn = in_sizes[0] / 57;

  // carve workspace (256B aligned)
  uintptr_t p = (uintptr_t)d_ws;
  auto carve = [&](size_t bytes) {
    uintptr_t r = p;
    p = (p + bytes + 255) & ~(uintptr_t)255;
    return (void*)r;
  };
  float* h_ws = (float*)carve((size_t)Nn * 128 * sizeof(float));
  float* x_ws = (float*)carve((size_t)Nn * 3 * sizeof(float));
  _Float16* h16 = (_Float16*)carve((size_t)Nn * 128 * sizeof(_Float16));
  float* h_nb = (float*)carve((size_t)Nn * 128 * sizeof(float));
  float* x_nb = (float*)carve((size_t)Nn * 3 * sizeof(float));
  _Float16* We1T = (_Float16*)carve((size_t)128 * 288 * 2);
  _Float16* We2T = (_Float16*)carve((size_t)128 * 128 * 2);
  _Float16* Wc1T = (_Float16*)carve((size_t)128 * 128 * 2);
  _Float16* Wn1T = (_Float16*)carve((size_t)128 * 256 * 2);
  _Float16* Wn2T = (_Float16*)carve((size_t)128 * 128 * 2);

  auto W = [&](int layer, int slot) -> const float* {
    return (const float*)d_in[4 + layer * 11 + slot];
  };

  const int TB = 256;
  auto g1 = [&](int n) {
    int b = (n + TB - 1) / TB;
    return b > 4096 ? 4096 : b;
  };

  for (int L = 0; L < 3; ++L) {
    const int IN = (L == 0) ? 57 : 128;
    const int K1 = 2 * IN + 1;             // edge MLP input width
    const int KPE = (L == 0) ? 128 : 288;  // K1 padded to 32
    const int KN = IN + 128;               // node MLP input width
    const int KPN = (L == 0) ? 192 : 256;  // KN padded to 32
    const float* hcur = (L == 0) ? h0 : h_ws;
    const float* xcur = (L == 0) ? x0 : x_ws;

    // f16 node features + transposed f16 weights
    k_f32_to_f16<<<g1(Nn * IN), TB, 0, stream>>>(hcur, h16, Nn * IN);
    k_transpose_w<<<g1(128 * KPE), TB, 0, stream>>>(W(L, 0), We1T, K1, KPE);
    k_transpose_w<<<g1(128 * 128), TB, 0, stream>>>(W(L, 2), We2T, 128, 128);
    k_transpose_w<<<g1(128 * 128), TB, 0, stream>>>(W(L, 4), Wc1T, 128, 128);
    k_transpose_w<<<g1(128 * KPN), TB, 0, stream>>>(W(L, 7), Wn1T, KN, KPN);
    k_transpose_w<<<g1(128 * 128), TB, 0, stream>>>(W(L, 9), Wn2T, 128, 128);

    // zero scatter accumulators
    k_zero_f32<<<g1(Nn * 128), TB, 0, stream>>>(h_nb, Nn * 128);
    k_zero_f32<<<g1(Nn * 3), TB, 0, stream>>>(x_nb, Nn * 3);

    if (L == 0) {
      k_egnn_edge<57, 128><<<1024, 128, 0, stream>>>(
          h16, xcur, src, dst, E, We1T, W(L, 1), We2T, W(L, 3), Wc1T, W(L, 5),
          W(L, 6), h_nb, x_nb);
      k_egnn_node<57, 192><<<512, 128, 0, stream>>>(
          hcur, h_nb, Wn1T, W(L, 8), Wn2T, W(L, 10), h_ws, Nn);
    } else {
      k_egnn_edge<128, 288><<<1024, 128, 0, stream>>>(
          h16, xcur, src, dst, E, We1T, W(L, 1), We2T, W(L, 3), Wc1T, W(L, 5),
          W(L, 6), h_nb, x_nb);
      k_egnn_node<128, 256><<<512, 128, 0, stream>>>(
          h_ws, h_nb, Wn1T, W(L, 8), Wn2T, W(L, 10), h_ws, Nn);
    }
    k_add_f32<<<g1(Nn * 3), TB, 0, stream>>>(xcur, x_nb, x_ws, Nn * 3);
  }

  k_mean<<<out_size, 256, 0, stream>>>(h_ws, x_ws, (float*)d_out, Nn);
}
